// differentiableSuperpixelTokenizer_51625506898388
// MI455X (gfx1250) — matrix operand order repaired and checked by reading, not compiled, not run on hardware
//
#include <hip/hip_runtime.h>
#include <hip/hip_bf16.h>

// ---------------------------------------------------------------------------
// Types / WMMA helpers (CDNA5 / gfx1250, wave32)
// ---------------------------------------------------------------------------
typedef __attribute__((ext_vector_type(16))) __bf16 v16bf;
typedef __attribute__((ext_vector_type(8)))  __bf16 v8bf;
typedef __attribute__((ext_vector_type(8)))  float  v8f;

#define DEV __device__ __forceinline__

DEV __bf16 to_bf(float x){ return (__bf16)x; }
DEV float  bf2f(__bf16 x){ return (float)x; }

DEV v16bf cat8(v8bf a, v8bf b){
  return __builtin_shufflevector(a, b, 0,1,2,3,4,5,6,7,8,9,10,11,12,13,14,15);
}

// A fragment: 16x32 bf16 tile, row-major, K contiguous.
DEV v16bf frag_a(const __bf16* base, int lda, int kk, int lane){
  int m = lane & 15, g = lane >> 4;
  const __bf16* p = base + (size_t)m * lda + g*8 + kk;
  v8bf a0 = *(const v8bf*)p;        // K = kk + g*8 + [0..7]
  v8bf a1 = *(const v8bf*)(p + 16); // K = kk + 16 + g*8 + [0..7]
  return cat8(a0, a1);
}
// B fragment: 32x16 bf16 tile; B stored as [N][K] (K contiguous).
DEV v16bf frag_b(const __bf16* base, int ldb, int kk, int lane){
  int n = lane & 15, g = lane >> 4;
  const __bf16* p = base + (size_t)n * ldb + g*8 + kk;
  v8bf b0 = *(const v8bf*)p;
  v8bf b1 = *(const v8bf*)(p + 16);
  return cat8(b0, b1);
}
DEV v8f wmma_bf(v16bf a, v16bf b, v8f c){
  return __builtin_amdgcn_wmma_f32_16x16x32_bf16(false, a, false, b, (short)0, c,
                                                 false, false);
}

// CDNA5 async global->LDS copy (ASYNCcnt path) + wait
DEV void async_ld_b128(unsigned lds_off, const void* gptr){
  asm volatile("global_load_async_to_lds_b128 %0, %1, off"
               :: "v"(lds_off), "v"(gptr) : "memory");
}
DEV void wait_async0(){ asm volatile("s_wait_asynccnt 0x0" ::: "memory"); }

// ---------------------------------------------------------------------------
// conv1: 7x7 s2 p3, 3->64, direct
// ---------------------------------------------------------------------------
__global__ void conv1_kernel(const float* __restrict__ img, const float* __restrict__ w,
                             const float* __restrict__ bias, float* __restrict__ out){
  int t = blockIdx.x * blockDim.x + threadIdx.x;
  if (t >= 2*64*112*112) return;
  int x = t % 112, y = (t/112) % 112, oc = (t/12544) % 64, b = t / (12544*64);
  float acc = bias[oc];
  for (int ic = 0; ic < 3; ++ic)
    for (int ky = 0; ky < 7; ++ky){
      int iy = 2*y + ky - 3; if ((unsigned)iy >= 224u) continue;
      for (int kx = 0; kx < 7; ++kx){
        int ix = 2*x + kx - 3; if ((unsigned)ix >= 224u) continue;
        acc += img[((b*3+ic)*224+iy)*224+ix] * w[((oc*3+ic)*7+ky)*7+kx];
      }
    }
  out[t] = acc; // [b][oc][y][x]
}

// ---------------------------------------------------------------------------
// BatchNorm batch statistics (training mode): mean + rsqrt(var+eps)
// ---------------------------------------------------------------------------
__global__ void bnstat_cf(const float* __restrict__ x, float* mean, float* inv,
                          int C, int perB, int Bn){
  int c = blockIdx.x, tid = threadIdx.x;
  __shared__ float s1[256], s2[256];
  float a = 0.f, q = 0.f;
  int total = Bn * perB;
  for (int i = tid; i < total; i += 256){
    int bb = i / perB, j = i - bb*perB;
    float v = x[((size_t)bb*C + c)*perB + j];
    a += v; q += v*v;
  }
  s1[tid] = a; s2[tid] = q; __syncthreads();
  for (int s = 128; s > 0; s >>= 1){
    if (tid < s){ s1[tid] += s1[tid+s]; s2[tid] += s2[tid+s]; }
    __syncthreads();
  }
  if (tid == 0){
    float m = s1[0] / total, v = s2[0] / total - m*m;
    mean[c] = m; inv[c] = rsqrtf(v + 1e-5f);
  }
}
__global__ void bnstat_cl(const float* __restrict__ x, float* mean, float* inv,
                          int C, int rows){
  int c = blockIdx.x, tid = threadIdx.x;
  __shared__ float s1[256], s2[256];
  float a = 0.f, q = 0.f;
  for (int i = tid; i < rows; i += 256){
    float v = x[(size_t)i*C + c];
    a += v; q += v*v;
  }
  s1[tid] = a; s2[tid] = q; __syncthreads();
  for (int s = 128; s > 0; s >>= 1){
    if (tid < s){ s1[tid] += s1[tid+s]; s2[tid] += s2[tid+s]; }
    __syncthreads();
  }
  if (tid == 0){
    float m = s1[0] / rows, v = s2[0] / rows - m*m;
    mean[c] = m; inv[c] = rsqrtf(v + 1e-5f);
  }
}

__global__ void bn_relu_cf(float* __restrict__ x, const float* mean, const float* inv,
                           const float* g, const float* b, int C, int perB, int total){
  int t = blockIdx.x * blockDim.x + threadIdx.x;
  if (t >= total) return;
  int c = (t / perB) % C;
  float v = (x[t] - mean[c]) * inv[c] * g[c] + b[c];
  x[t] = v > 0.f ? v : 0.f;
}

// ---------------------------------------------------------------------------
// im2col for conv2 (3x3 p1): patches [M=25088][K=576] bf16, K order = c*9+ky*3+kx
// ---------------------------------------------------------------------------
__global__ void im2col_kernel(const float* __restrict__ act1, __bf16* __restrict__ patch){
  long long t = (long long)blockIdx.x * blockDim.x + threadIdx.x;
  if (t >= (long long)25088*576) return;
  int m = (int)(t / 576), kidx = (int)(t - (long long)m*576);
  int b = m / 12544, p = m - b*12544;
  int y = p / 112, x = p - y*112;
  int cc = kidx / 9, r = kidx - cc*9, ky = r/3, kx = r - ky*3;
  int yy = y + ky - 1, xx = x + kx - 1;
  float v = 0.f;
  if ((unsigned)yy < 112u && (unsigned)xx < 112u)
    v = act1[(((size_t)b*64 + cc)*112 + yy)*112 + xx];
  patch[t] = to_bf(v);
}

// ---------------------------------------------------------------------------
// f32 -> bf16 conversion (weights)
// ---------------------------------------------------------------------------
__global__ void cvt_f32_bf16(const float* __restrict__ src, __bf16* __restrict__ dst, int n){
  int t = blockIdx.x * blockDim.x + threadIdx.x;
  if (t < n) dst[t] = to_bf(src[t]);
}

// ---------------------------------------------------------------------------
// Generic NT GEMM, bf16 in, f32 accumulate, double-buffered fragment loads so
// next step's global_load_b128s are in flight while the WMMA executes.
// mode 0: write f32 (Cf) and/or bf16 (Cb); mode 1: qkv split (v transposed).
// ---------------------------------------------------------------------------
__global__ void gemm_bf16_nt(const __bf16* __restrict__ A, int lda,
                             const __bf16* __restrict__ Bm, int ldb,
                             const float* __restrict__ bias,
                             int N, int Ktot, int mode,
                             float* __restrict__ Cf, __bf16* __restrict__ Cb,
                             __bf16* __restrict__ qb, __bf16* __restrict__ kb,
                             __bf16* __restrict__ vT){
  int lane = threadIdx.x & 31, wid = threadIdx.x >> 5;
  int row0 = blockIdx.x * 16;
  int col0 = blockIdx.y * 128 + wid * 16;
  int m0 = lane & 15, g = lane >> 4;
  const __bf16* arow = A  + (size_t)(row0 + m0) * lda + g*8;
  const __bf16* brow = Bm + (size_t)(col0 + m0) * ldb + g*8;
  // preload step 0
  v8bf a0 = *(const v8bf*)(arow);
  v8bf a1 = *(const v8bf*)(arow + 16);
  v8bf b0 = *(const v8bf*)(brow);
  v8bf b1 = *(const v8bf*)(brow + 16);
  v8f c = {};
  for (int kk = 32; kk < Ktot; kk += 32){
    // issue next step's loads before the matrix op (latency hiding)
    v8bf na0 = *(const v8bf*)(arow + kk);
    v8bf na1 = *(const v8bf*)(arow + kk + 16);
    v8bf nb0 = *(const v8bf*)(brow + kk);
    v8bf nb1 = *(const v8bf*)(brow + kk + 16);
    if (kk + 96 < Ktot){
      __builtin_prefetch(arow + kk + 96, 0, 1);   // global_prefetch_b8
      __builtin_prefetch(brow + kk + 96, 0, 1);
    }
    c = wmma_bf(cat8(a0, a1), cat8(b0, b1), c);
    a0 = na0; a1 = na1; b0 = nb0; b1 = nb1;
  }
  c = wmma_bf(cat8(a0, a1), cat8(b0, b1), c);

  int n = col0 + m0;
  float bv = bias ? bias[n] : 0.f;
#pragma unroll
  for (int r = 0; r < 8; ++r){
    int m = row0 + g*8 + r;
    float v = c[r] + bv;
    if (mode == 0){
      if (Cf) Cf[(size_t)m * N + n] = v;
      if (Cb) Cb[(size_t)m * N + n] = to_bf(v);
    } else {
      if (n < 768)        qb[(size_t)m*768 + n]        = to_bf(v);
      else if (n < 1536)  kb[(size_t)m*768 + (n-768)]  = to_bf(v);
      else {
        int bs = m >> 7, kt = m & 127;                  // token -> (segment, slot)
        vT[((size_t)bs*768 + (n-1536))*128 + kt] = to_bf(v);  // V^T per segment
      }
    }
  }
}

// ---------------------------------------------------------------------------
// Segment resize + 9-tap equality mask (for positional embedding)
// ---------------------------------------------------------------------------
__global__ void seg_mask(const int* __restrict__ segments, int* __restrict__ segf,
                         unsigned* __restrict__ mask){
  int t = blockIdx.x * blockDim.x + threadIdx.x;
  if (t >= 2*12544) return;
  int b = t / 12544, p = t - b*12544;
  int y = p / 112, x = p - y*112;
  const int* sb = segments + b*224*224;
  int s = sb[(2*y)*224 + 2*x];
  segf[t] = s;
  unsigned m = 0;
  for (int ky = 0; ky < 3; ++ky){
    int ny = y + ky - 1; if ((unsigned)ny >= 112u) continue;
    for (int kx = 0; kx < 3; ++kx){
      int nx = x + kx - 1; if ((unsigned)nx >= 112u) continue;
      if (sb[(2*ny)*224 + 2*nx] == s) m |= 1u << (ky*3+kx);
    }
  }
  mask[t] = m;
}

// ---------------------------------------------------------------------------
// feats = bf16( relu(bn2(x2)) + pos_b + sum_taps pos_w )   [B*HW][768]
// ---------------------------------------------------------------------------
__global__ void feats_kernel(const float* __restrict__ x2, const float* mean,
                             const float* inv, const float* g, const float* bb,
                             const float* __restrict__ pos_w, const float* pos_b,
                             const unsigned* __restrict__ mask,
                             __bf16* __restrict__ feats){
  long long t = (long long)blockIdx.x * blockDim.x + threadIdx.x;
  if (t >= (long long)25088*768) return;
  int m = (int)(t / 768), c = (int)(t - (long long)m*768);
  float v = x2[t];
  v = (v - mean[c]) * inv[c] * g[c] + bb[c];
  v = v > 0.f ? v : 0.f;
  unsigned mk = mask[m];
  float pe = pos_b[c];
#pragma unroll
  for (int j = 0; j < 9; ++j)
    if (mk & (1u << j)) pe += pos_w[c*9 + j];
  feats[t] = to_bf(v + pe);
}

// ---------------------------------------------------------------------------
// Deterministic per-segment index build: one wave per (b,s), ballot+popcount scan
// ---------------------------------------------------------------------------
__global__ void build_idx(const int* __restrict__ segf, int* __restrict__ idxs,
                          int* __restrict__ counts){
  int bs = blockIdx.x;             // 0..391
  int b = bs / 196, s = bs - b*196;
  int lane = threadIdx.x;          // 32 threads
  int* myidx = idxs + bs*128;
  for (int j = lane; j < 128; j += 32) myidx[j] = 12544;   // sentinel
  const int* sf = segf + b*12544;
  int base = 0;
  for (int it = 0; it < 392; ++it){
    int p = it*32 + lane;
    bool pred = (sf[p] == s);
    unsigned mb = (unsigned)__ballot(pred);
    if (pred){
      int r = base + __popc(mb & ((1u << lane) - 1u));
      if (r < 128) myidx[r] = p;
    }
    base += __popc(mb);
  }
  if (lane == 0) counts[bs] = base < 128 ? base : 128;
}

// ---------------------------------------------------------------------------
// Gather tokens [B*S*128][768] bf16 (16B vectors; padding slots -> 0)
// ---------------------------------------------------------------------------
__global__ void gather_tokens(const __bf16* __restrict__ feats, const int* __restrict__ idxs,
                              __bf16* __restrict__ tokens){
  int t = blockIdx.x * blockDim.x + threadIdx.x;   // one v8bf (8 channels) each
  if (t >= 50176*96) return;
  int tok = t / 96, cblk = t - tok*96;
  int c0 = cblk * 8;
  int bs = tok >> 7, kk = tok & 127;
  int idx = idxs[bs*128 + kk];
  v8bf v = {};
  if (idx < 12544){
    int b = bs / 196;
    v = *(const v8bf*)(feats + ((size_t)b*12544 + idx)*768 + c0);
  }
  *(v8bf*)(tokens + (size_t)tok*768 + c0) = v;
}

// ---------------------------------------------------------------------------
// Per-(segment,head) attention.
//  - async-DMA V^T slab (96x128 bf16, 24KB) into LDS, overlapped with scores
//  - scores via WMMA (K=96), masked softmax, attn*V via WMMA (A and B from LDS)
// Block = 128 threads (4 waves). LDS: 64KB scores + 36KB attn + 24KB V tile.
// ---------------------------------------------------------------------------
__global__ void attn_kernel(const __bf16* __restrict__ q, const __bf16* __restrict__ k,
                            const __bf16* __restrict__ vT, const int* __restrict__ counts,
                            __bf16* __restrict__ o_mid){
  __shared__ float  sc[128*128];
  __shared__ __bf16 abf[128*144];
  __shared__ __align__(16) __bf16 vtile[96*128];
  int bs = blockIdx.x >> 3, h = blockIdx.x & 7;
  int tid = threadIdx.x;
  int lane = tid & 31, wid = tid >> 5;
  int g = lane >> 4, m0 = lane & 15;
  const __bf16* qh = q + (size_t)bs*128*768 + h*96;
  const __bf16* kh = k + (size_t)bs*128*768 + h*96;
  const __bf16* vh = vT + ((size_t)bs*768 + h*96)*128;   // contiguous 96x128 slab
  const float scale = 0.1020620726f; // 96^-0.5

  // kick off async DMA of V^T slab into LDS (12 passes x 128 threads x 16B = 24KB)
  {
    unsigned dst = (unsigned)(size_t)(void*)vtile;       // LDS byte offset
    const char* src = (const char*)vh;
#pragma unroll
    for (int pass = 0; pass < 12; ++pass){
      int off = pass*2048 + tid*16;
      async_ld_b128(dst + (unsigned)off, src + off);
    }
  }

  // scores: 8x8 tiles of 16x16, K=96 (overlaps with the async DMA above)
  for (int i = 0; i < 16; ++i){
    int t = wid*16 + i, tm = t >> 3, tn = t & 7;
    const __bf16* Ap = qh + (size_t)(tm*16)*768;
    const __bf16* Bp = kh + (size_t)(tn*16)*768;
    v16bf af0 = frag_a(Ap, 768, 0,  lane);
    v16bf af1 = frag_a(Ap, 768, 32, lane);
    v16bf af2 = frag_a(Ap, 768, 64, lane);
    v16bf bf0 = frag_b(Bp, 768, 0,  lane);
    v16bf bf1 = frag_b(Bp, 768, 32, lane);
    v16bf bf2 = frag_b(Bp, 768, 64, lane);
    v8f c = {};
    c = wmma_bf(af0, bf0, c);
    c = wmma_bf(af1, bf1, c);
    c = wmma_bf(af2, bf2, c);
    int n = tn*16 + m0;
#pragma unroll
    for (int r = 0; r < 8; ++r){
      int m = tm*16 + g*8 + r;
      sc[m*128 + n] = c[r] * scale;
    }
  }
  __syncthreads();
  int cnt = counts[bs];
  // masked softmax, one wave per row (rows wid, wid+4, ...)
  for (int i = 0; i < 32; ++i){
    int row = wid + 4*i;
    float v[4];
#pragma unroll
    for (int j = 0; j < 4; ++j){
      int col = lane + 32*j;
      float x = sc[row*128 + col];
      v[j] = (col < cnt) ? x : -1e9f;
    }
    float mx = fmaxf(fmaxf(v[0], v[1]), fmaxf(v[2], v[3]));
#pragma unroll
    for (int d2 = 16; d2 > 0; d2 >>= 1) mx = fmaxf(mx, __shfl_xor(mx, d2));
    float s = 0.f;
#pragma unroll
    for (int j = 0; j < 4; ++j){ v[j] = __expf(v[j] - mx); s += v[j]; }
#pragma unroll
    for (int d2 = 16; d2 > 0; d2 >>= 1) s += __shfl_xor(s, d2);
    float invs = 1.f / s;
#pragma unroll
    for (int j = 0; j < 4; ++j){
      int col = lane + 32*j;
      abf[row*144 + col] = to_bf(v[j] * invs);
    }
  }
  wait_async0();          // V^T slab resident in LDS
  __syncthreads();
  // o = attn * V_h : 8x6 tiles of 16x16, K=128; both operands from LDS
  for (int i = 0; i < 12; ++i){
    int t = wid*12 + i, tm = t/6, tn = t - tm*6;
    const __bf16* Ap = (const __bf16*)abf + (size_t)(tm*16)*144;
    const __bf16* Bp = (const __bf16*)vtile + (size_t)(tn*16)*128;
    v8f c = {};
#pragma unroll
    for (int kk = 0; kk < 128; kk += 32)
      c = wmma_bf(frag_a(Ap, 144, kk, lane), frag_b(Bp, 128, kk, lane), c);
    int n = tn*16 + m0;
#pragma unroll
    for (int r = 0; r < 8; ++r){
      int m = tm*16 + g*8 + r;
      o_mid[((size_t)bs*128 + m)*768 + h*96 + n] = to_bf(c[r]);
    }
  }
}

// ---------------------------------------------------------------------------
// Masked mean pool over valid tokens -> [2,196,768] f32
// ---------------------------------------------------------------------------
__global__ void pool_kernel(const __bf16* __restrict__ oproj, const int* __restrict__ counts,
                            float* __restrict__ out){
  int t = blockIdx.x * blockDim.x + threadIdx.x;
  if (t >= 392*768) return;
  int bs = t / 768, d = t - bs*768;
  int cnt = counts[bs];
  float acc = 0.f;
  const __bf16* p = oproj + (size_t)bs*128*768 + d;
  for (int kk = 0; kk < cnt; ++kk) acc += bf2f(p[(size_t)kk*768]);
  out[t] = cnt ? acc / (float)cnt : 0.f;
}

// ---------------------------------------------------------------------------
// Host launcher
// ---------------------------------------------------------------------------
extern "C" void kernel_launch(void* const* d_in, const int* in_sizes, int n_in,
                              void* d_out, int out_size, void* d_ws, size_t ws_size,
                              hipStream_t stream) {
  (void)in_sizes; (void)n_in; (void)out_size; (void)ws_size;
  const float* img     = (const float*)d_in[0];
  const int*   segs    = (const int*)  d_in[1];
  const float* conv1_w = (const float*)d_in[2];
  const float* conv1_b = (const float*)d_in[3];
  const float* bn1_g   = (const float*)d_in[4];
  const float* bn1_b   = (const float*)d_in[5];
  const float* conv2_w = (const float*)d_in[6];
  const float* conv2_b = (const float*)d_in[7];
  const float* bn2_g   = (const float*)d_in[8];
  const float* bn2_b   = (const float*)d_in[9];
  const float* pos_w   = (const float*)d_in[10];
  const float* pos_b   = (const float*)d_in[11];
  const float* in_w    = (const float*)d_in[12];
  const float* in_b    = (const float*)d_in[13];
  const float* out_w   = (const float*)d_in[14];
  const float* out_b   = (const float*)d_in[15];
  float* out = (float*)d_out;

  // workspace layout
  char* w8 = (char*)d_ws;
  size_t off = 0;
  auto take = [&](size_t bytes) -> void* {
    void* p = w8 + off; off = (off + bytes + 255) & ~(size_t)255; return p;
  };
  float*    act1   = (float*)   take((size_t)1605632*4);   // [2][64][112][112]
  __bf16*   patch  = (__bf16*)  take((size_t)14450688*2);  // [25088][576]
  float*    x2     = (float*)   take((size_t)19267584*4);  // [25088][768]
  float*    mean1  = (float*)   take(64*4);
  float*    inv1   = (float*)   take(64*4);
  float*    mean2  = (float*)   take(768*4);
  float*    inv2   = (float*)   take(768*4);
  int*      segf   = (int*)     take(25088*4);
  unsigned* maskb  = (unsigned*)take(25088*4);
  __bf16*   featsb = (__bf16*)  take((size_t)19267584*2);  // [25088][768]
  __bf16*   w2b    = (__bf16*)  take((size_t)442368*2);    // [768][576]
  __bf16*   inwb   = (__bf16*)  take((size_t)1769472*2);   // [2304][768]
  __bf16*   outwb  = (__bf16*)  take((size_t)589824*2);    // [768][768]
  int*      idxs   = (int*)     take(50176*4);             // [392][128]
  int*      counts = (int*)     take(392*4);
  __bf16*   tokens = (__bf16*)  take((size_t)38535168*2);  // [50176][768]
  __bf16*   qb     = (__bf16*)  take((size_t)38535168*2);
  __bf16*   kb     = (__bf16*)  take((size_t)38535168*2);
  __bf16*   vT     = (__bf16*)  take((size_t)38535168*2);  // [392][768][128]
  __bf16*   omid   = (__bf16*)  take((size_t)38535168*2);
  __bf16*   oproj  = (__bf16*)  take((size_t)38535168*2);

  auto nb = [](long long n){ return (unsigned)((n + 255) / 256); };

  // weight conversions
  cvt_f32_bf16<<<nb(442368), 256, 0, stream>>>(conv2_w, w2b, 442368);
  cvt_f32_bf16<<<nb(1769472), 256, 0, stream>>>(in_w, inwb, 1769472);
  cvt_f32_bf16<<<nb(589824), 256, 0, stream>>>(out_w, outwb, 589824);

  // conv1 + BN1(train) + ReLU
  conv1_kernel<<<nb(1605632), 256, 0, stream>>>(img, conv1_w, conv1_b, act1);
  bnstat_cf<<<64, 256, 0, stream>>>(act1, mean1, inv1, 64, 12544, 2);
  bn_relu_cf<<<nb(1605632), 256, 0, stream>>>(act1, mean1, inv1, bn1_g, bn1_b,
                                              64, 12544, 1605632);

  // conv2 as implicit GEMM (WMMA): [25088 x 768] = patches[25088x576] * w2b^T
  im2col_kernel<<<nb((long long)25088*576), 256, 0, stream>>>(act1, patch);
  gemm_bf16_nt<<<dim3(25088/16, 768/128), 256, 0, stream>>>(
      patch, 576, w2b, 576, conv2_b, 768, 576, 0, x2, nullptr,
      nullptr, nullptr, nullptr);
  bnstat_cl<<<768, 256, 0, stream>>>(x2, mean2, inv2, 768, 25088);

  // positional-embedding mask + feats
  seg_mask<<<nb(25088), 256, 0, stream>>>(segs, segf, maskb);
  feats_kernel<<<nb((long long)25088*768), 256, 0, stream>>>(
      x2, mean2, inv2, bn2_g, bn2_b, pos_w, pos_b, maskb, featsb);

  // superpixel token build
  build_idx<<<392, 32, 0, stream>>>(segf, idxs, counts);
  gather_tokens<<<nb((long long)50176*96), 256, 0, stream>>>(featsb, idxs, tokens);

  // QKV projection (WMMA), V stored transposed per segment
  gemm_bf16_nt<<<dim3(50176/16, 2304/128), 256, 0, stream>>>(
      tokens, 768, inwb, 768, in_b, 2304, 768, 1, nullptr, nullptr,
      qb, kb, vT);

  // attention per (segment, head)
  attn_kernel<<<392*8, 128, 0, stream>>>(qb, kb, vT, counts, omid);

  // output projection (WMMA) + masked mean pool
  gemm_bf16_nt<<<dim3(50176/16, 768/128), 256, 0, stream>>>(
      omid, 768, outwb, 768, out_b, 768, 768, 0, nullptr, oproj,
      nullptr, nullptr, nullptr);
  pool_kernel<<<nb(392*768), 256, 0, stream>>>(oproj, counts, out);
}
